// ConformerBlock_6493990552072
// MI455X (gfx1250) — compile-verified, matrix operand order checked by reading, not run on hardware
//
#include <hip/hip_runtime.h>
#include <hip/hip_bf16.h>

// ---------------- CDNA5 WMMA types ----------------
typedef __attribute__((ext_vector_type(16))) __bf16 bfrag16;
typedef __attribute__((ext_vector_type(8)))  float  v8f;

#define D_MODEL 512
#define N_HEADS 8
#define DKH     64
#define D_FF    2048
#define KSIZE   31
#define BATCH   8
#define SEQ     1024
#define NTOK    (BATCH * SEQ)
#define EPSF    1e-5f

union FragU { bfrag16 v; uint4 q[2]; };

// Load a 16x32 bf16 WMMA A/B fragment from a k-contiguous buffer.
// ISA 16-bit layout: lane = half*16 + r; elements 0..7 <- k = half*8+0..7,
// elements 8..15 <- k = 16+half*8+0..7.  Two b128 loads per lane.
__device__ __forceinline__ bfrag16 load_frag(const __bf16* base, int ldk) {
    const int lane = threadIdx.x & 31;
    const int half = lane >> 4;
    const int r    = lane & 15;
    const __bf16* p = base + (size_t)r * ldk + half * 8;
    FragU f;
    f.q[0] = *(const uint4*)(p);
    f.q[1] = *(const uint4*)(p + 16);
    return f.v;
}

// CDNA5 async memory->LDS copy (16B per lane), tracked by ASYNCcnt.
// LDS flat addresses carry the LDS byte offset in the low 32 bits.
__device__ __forceinline__ void async_copy_b128(const void* gptr, void* lptr) {
    unsigned lds = (unsigned)(uintptr_t)lptr;
    asm volatile("global_load_async_to_lds_b128 %0, %1, off"
                 :: "v"(lds), "v"(gptr) : "memory");
}
__device__ __forceinline__ void wait_async0() {
    asm volatile("s_wait_asynccnt 0x0" ::: "memory");
}

__device__ __forceinline__ float silu_f(float v) { return v / (1.f + __expf(-v)); }

// ---------------- Tiled bf16 GEMM with fused epilogues ----------------
// C[M,N] = epilogue(A[M,K](bf16, row-major) x Bt[N,K](bf16, pre-transposed) + bias)
// EPI 0: f32 = acc+bias            (pw1, pre-GLU)
// EPI 1: bf16 = silu(acc+bias)     (FFN mid)
// EPI 2: f32 = resid + 0.5*(acc+b) (FFN out)
// EPI 3: bf16 = acc+bias           (Q/K/V proj)
// EPI 4: f32 = resid + (acc+bias)  (attn out proj / pw2)
template<int EPI>
__global__ __launch_bounds__(256)
void gemm_bf16(const __bf16* __restrict__ A, const __bf16* __restrict__ Bt,
               const float* __restrict__ bias, const float* __restrict__ resid,
               float* __restrict__ outf, __bf16* __restrict__ outb,
               int M, int N, int K)
{
    // Double-buffered k-contiguous tiles; rows padded to 40 bf16 (80B):
    // 16B-aligned chunks, 20-bank row stride (no fragment-load conflicts).
    __shared__ __bf16 As[2][128 * 40];
    __shared__ __bf16 Bs[2][128 * 40];

    const int tid  = threadIdx.x;
    const int lane = tid & 31;
    const int wave = tid >> 5;
    const int wm   = wave & 3;   // 4 row-strips of 32
    const int wn   = wave >> 2;  // 2 col-strips of 64
    const int m0   = blockIdx.y * 128;
    const int n0   = blockIdx.x * 128;

    v8f acc[2][4];
    #pragma unroll
    for (int i = 0; i < 2; i++)
        #pragma unroll
        for (int j = 0; j < 4; j++)
            #pragma unroll
            for (int e = 0; e < 8; e++) acc[i][j][e] = 0.f;

    // Each tile (128 rows x 32 bf16) = 512 16B chunks; thread owns chunks
    // tid and tid+256 of both the A and B tile -> 4 async copies per step.
    auto stage = [&](int buf, int kb) {
        #pragma unroll
        for (int i = 0; i < 2; i++) {
            const int c   = tid + i * 256;
            const int row = c >> 2;
            const int kc  = (c & 3) * 8;
            async_copy_b128(A  + (size_t)(m0 + row) * K + kb + kc,
                            &As[buf][row * 40 + kc]);
            async_copy_b128(Bt + (size_t)(n0 + row) * K + kb + kc,
                            &Bs[buf][row * 40 + kc]);
        }
    };

    stage(0, 0);
    wait_async0();
    __syncthreads();

    const int nsteps = K >> 5;
    for (int step = 0; step < nsteps; step++) {
        const int cur = step & 1;
        if (step + 1 < nsteps) stage(cur ^ 1, (step + 1) << 5);

        bfrag16 af[2], bf[4];
        af[0] = load_frag(&As[cur][(wm * 32 + 0)  * 40], 40);
        af[1] = load_frag(&As[cur][(wm * 32 + 16) * 40], 40);
        #pragma unroll
        for (int ni = 0; ni < 4; ni++)
            bf[ni] = load_frag(&Bs[cur][(wn * 64 + ni * 16) * 40], 40);

        #pragma unroll
        for (int mi = 0; mi < 2; mi++)
            #pragma unroll
            for (int ni = 0; ni < 4; ni++)
                acc[mi][ni] = __builtin_amdgcn_wmma_f32_16x16x32_bf16(
                    false, af[mi], false, bf[ni], (short)0, acc[mi][ni], false, false);

        wait_async0();      // own prefetch landed
        __syncthreads();    // everyone's prefetch landed, everyone's reads drained
    }

    const int colL = lane & 15;
    const int half = lane >> 4;
    #pragma unroll
    for (int mi = 0; mi < 2; mi++) {
        #pragma unroll
        for (int ni = 0; ni < 4; ni++) {
            const int gn = n0 + wn * 64 + ni * 16 + colL;
            const float bia = bias[gn];
            #pragma unroll
            for (int e = 0; e < 8; e++) {
                const int gm = m0 + wm * 32 + mi * 16 + e + half * 8;
                const size_t idx = (size_t)gm * N + gn;
                const float v = acc[mi][ni][e] + bia;
                if (EPI == 0)      outf[idx] = v;
                else if (EPI == 1) outb[idx] = (__bf16)silu_f(v);
                else if (EPI == 2) outf[idx] = resid[idx] + 0.5f * v;
                else if (EPI == 3) outb[idx] = (__bf16)v;
                else               outf[idx] = resid[idx] + v;
            }
        }
    }
}

// ---------------- Flash attention: one wave per (b, h, 16-query tile) ----------------
__global__ __launch_bounds__(32)
void attention_k(const __bf16* __restrict__ Q, const __bf16* __restrict__ Kb,
                 const __bf16* __restrict__ Vt, __bf16* __restrict__ ctx)
{
    __shared__ __bf16 P[16 * 32];  // probability tile, C-layout -> A-layout bounce
    const int lane = threadIdx.x & 31;
    const int half = lane >> 4;
    const int colL = lane & 15;
    const int q0 = blockIdx.x * 16;
    const int h  = blockIdx.y;
    const int b  = blockIdx.z;
    const float scl = 0.125f;  // 1/sqrt(64)

    const __bf16* qbase = Q + ((size_t)(b * SEQ + q0)) * D_MODEL + h * DKH;
    bfrag16 qa0 = load_frag(qbase, D_MODEL);
    bfrag16 qa1 = load_frag(qbase + 32, D_MODEL);

    float rmax[8], rsum[8];
    v8f ctxa[4];
    #pragma unroll
    for (int e = 0; e < 8; e++) { rmax[e] = -1e30f; rsum[e] = 0.f; }
    #pragma unroll
    for (int ni = 0; ni < 4; ni++)
        #pragma unroll
        for (int e = 0; e < 8; e++) ctxa[ni][e] = 0.f;

    for (int c = 0; c < SEQ; c += 32) {
        v8f s0, s1;
        #pragma unroll
        for (int e = 0; e < 8; e++) { s0[e] = 0.f; s1[e] = 0.f; }

        const __bf16* kbase = Kb + ((size_t)(b * SEQ + c)) * D_MODEL + h * DKH;
        bfrag16 k00 = load_frag(kbase, D_MODEL);
        bfrag16 k01 = load_frag(kbase + 32, D_MODEL);
        bfrag16 k10 = load_frag(kbase + 16 * D_MODEL, D_MODEL);
        bfrag16 k11 = load_frag(kbase + 16 * D_MODEL + 32, D_MODEL);

        s0 = __builtin_amdgcn_wmma_f32_16x16x32_bf16(false, qa0, false, k00, (short)0, s0, false, false);
        s0 = __builtin_amdgcn_wmma_f32_16x16x32_bf16(false, qa1, false, k01, (short)0, s0, false, false);
        s1 = __builtin_amdgcn_wmma_f32_16x16x32_bf16(false, qa0, false, k10, (short)0, s1, false, false);
        s1 = __builtin_amdgcn_wmma_f32_16x16x32_bf16(false, qa1, false, k11, (short)0, s1, false, false);

        // online softmax per row (rows live in 16-lane half-groups)
        #pragma unroll
        for (int e = 0; e < 8; e++) {
            float v0 = s0[e] * scl, v1 = s1[e] * scl;
            float m = fmaxf(v0, v1);
            m = fmaxf(m, __shfl_xor(m, 1));
            m = fmaxf(m, __shfl_xor(m, 2));
            m = fmaxf(m, __shfl_xor(m, 4));
            m = fmaxf(m, __shfl_xor(m, 8));
            float mn   = fmaxf(rmax[e], m);
            float corr = __expf(rmax[e] - mn);
            float p0 = __expf(v0 - mn), p1 = __expf(v1 - mn);
            float ps = p0 + p1;
            ps += __shfl_xor(ps, 1);
            ps += __shfl_xor(ps, 2);
            ps += __shfl_xor(ps, 4);
            ps += __shfl_xor(ps, 8);
            rsum[e] = rsum[e] * corr + ps;
            rmax[e] = mn;
            #pragma unroll
            for (int ni = 0; ni < 4; ni++) ctxa[ni][e] *= corr;
            const int row = e + half * 8;
            P[row * 32 + colL]      = (__bf16)p0;
            P[row * 32 + colL + 16] = (__bf16)p1;
        }
        __syncthreads();

        bfrag16 pf = load_frag(P, 32);
        const __bf16* vbase = Vt + ((size_t)(b * N_HEADS + h)) * DKH * SEQ + c;
        #pragma unroll
        for (int ni = 0; ni < 4; ni++) {
            bfrag16 vf = load_frag(vbase + (size_t)(ni * 16) * SEQ, SEQ);
            ctxa[ni] = __builtin_amdgcn_wmma_f32_16x16x32_bf16(
                false, pf, false, vf, (short)0, ctxa[ni], false, false);
        }
        __syncthreads();
    }

    #pragma unroll
    for (int ni = 0; ni < 4; ni++)
        #pragma unroll
        for (int e = 0; e < 8; e++) {
            const int row = q0 + e + half * 8;
            const int col = h * DKH + ni * 16 + colL;
            ctx[((size_t)(b * SEQ) + row) * D_MODEL + col] = (__bf16)(ctxa[ni][e] / rsum[e]);
        }
}

// ---------------- LayerNorm: one wave per token row ----------------
template<int OUTF>
__global__ __launch_bounds__(256)
void layernorm_k(const float* __restrict__ x, const float* __restrict__ g,
                 const float* __restrict__ bta, __bf16* __restrict__ ob,
                 float* __restrict__ of)
{
    const int lane = threadIdx.x & 31;
    const int row  = blockIdx.x * 8 + (threadIdx.x >> 5);
    const float* xr = x + (size_t)row * D_MODEL;
    float v[16];
    float s = 0.f, s2 = 0.f;
    #pragma unroll
    for (int i = 0; i < 16; i++) {
        float t = xr[lane + i * 32];
        v[i] = t; s += t; s2 += t * t;
    }
    #pragma unroll
    for (int off = 16; off >= 1; off >>= 1) {
        s  += __shfl_xor(s, off);
        s2 += __shfl_xor(s2, off);
    }
    const float mean = s * (1.f / D_MODEL);
    const float var  = s2 * (1.f / D_MODEL) - mean * mean;
    const float rstd = rsqrtf(var + EPSF);
    #pragma unroll
    for (int i = 0; i < 16; i++) {
        const int c = lane + i * 32;
        const float o = (v[i] - mean) * rstd * g[c] + bta[c];
        if (OUTF) of[(size_t)row * D_MODEL + c] = o;
        else      ob[(size_t)row * D_MODEL + c] = (__bf16)o;
    }
}

// ---------------- elementwise helpers ----------------
// f32 [K,N] -> bf16 transposed [N,K] (k-contiguous for async tile copies)
__global__ void f2bf_t_k(const float* __restrict__ s, __bf16* __restrict__ d,
                         int K, int N) {
    int idx = blockIdx.x * 256 + threadIdx.x;
    if (idx >= K * N) return;
    int k = idx / N, n = idx - k * N;
    d[(size_t)n * K + k] = (__bf16)s[idx];
}

__global__ void glu_k(const float* __restrict__ p, float* __restrict__ o) {
    int idx = blockIdx.x * 256 + threadIdx.x;  // over NTOK*512
    int c = idx & (D_MODEL - 1);
    int n = idx >> 9;
    float a  = p[(size_t)n * (2 * D_MODEL) + c];
    float gg = p[(size_t)n * (2 * D_MODEL) + D_MODEL + c];
    o[idx] = a / (1.f + __expf(-gg));
}

__global__ void transpose_v_k(const __bf16* __restrict__ V, __bf16* __restrict__ Vt) {
    int idx = blockIdx.x * 256 + threadIdx.x;  // over NTOK*512
    int c = idx & (D_MODEL - 1);
    int n = idx >> 9;
    int s = n & (SEQ - 1);
    int b = n >> 10;
    int h = c >> 6;
    int d = c & (DKH - 1);
    Vt[(((size_t)(b * N_HEADS + h)) * DKH + d) * SEQ + s] = V[(size_t)idx];
}

__global__ void dwconv_k(const float* __restrict__ gin, const float* __restrict__ w,
                         const float* __restrict__ wb, const float* __restrict__ bng,
                         const float* __restrict__ bnb, const float* __restrict__ bnm,
                         const float* __restrict__ bnv, __bf16* __restrict__ out) {
    int idx = blockIdx.x * 256 + threadIdx.x;  // over NTOK*512
    int d = idx & (D_MODEL - 1);
    int n = idx >> 9;
    int s = n & (SEQ - 1);
    int b = n >> 10;
    float acc = 0.f;
    #pragma unroll
    for (int t = 0; t < KSIZE; t++) {
        int ss = s + t - (KSIZE / 2);
        if (ss >= 0 && ss < SEQ)
            acc += gin[((size_t)(b * SEQ + ss)) * D_MODEL + d] * w[d * KSIZE + t];
    }
    acc += wb[d];
    acc = (acc - bnm[d]) * rsqrtf(bnv[d] + EPSF) * bng[d] + bnb[d];
    out[idx] = (__bf16)silu_f(acc);
}

// ---------------- host orchestration ----------------
extern "C" void kernel_launch(void* const* d_in, const int* in_sizes, int n_in,
                              void* d_out, int out_size, void* d_ws, size_t ws_size,
                              hipStream_t stream) {
    const float* x        = (const float*)d_in[0];
    const float* f1_ln_g  = (const float*)d_in[1];
    const float* f1_ln_b  = (const float*)d_in[2];
    const float* f1_w1    = (const float*)d_in[3];
    const float* f1_b1    = (const float*)d_in[4];
    const float* f1_w2    = (const float*)d_in[5];
    const float* f1_b2    = (const float*)d_in[6];
    const float* mh_ln_g  = (const float*)d_in[7];
    const float* mh_ln_b  = (const float*)d_in[8];
    const float* wq       = (const float*)d_in[9];
    const float* bq       = (const float*)d_in[10];
    const float* wk       = (const float*)d_in[11];
    const float* bk       = (const float*)d_in[12];
    const float* wv       = (const float*)d_in[13];
    const float* bv       = (const float*)d_in[14];
    const float* wo       = (const float*)d_in[15];
    const float* bo       = (const float*)d_in[16];
    const float* cv_ln_g  = (const float*)d_in[17];
    const float* cv_ln_b  = (const float*)d_in[18];
    const float* pw1_w    = (const float*)d_in[19];
    const float* pw1_b    = (const float*)d_in[20];
    const float* dw_w     = (const float*)d_in[21];
    const float* dw_b     = (const float*)d_in[22];
    const float* bn_g     = (const float*)d_in[23];
    const float* bn_b     = (const float*)d_in[24];
    const float* bn_mean  = (const float*)d_in[25];
    const float* bn_var   = (const float*)d_in[26];
    const float* pw2_w    = (const float*)d_in[27];
    const float* pw2_b    = (const float*)d_in[28];
    const float* f2_ln_g  = (const float*)d_in[29];
    const float* f2_ln_b  = (const float*)d_in[30];
    const float* f2_w1    = (const float*)d_in[31];
    const float* f2_b1    = (const float*)d_in[32];
    const float* f2_w2    = (const float*)d_in[33];
    const float* f2_b2    = (const float*)d_in[34];
    const float* fn_ln_g  = (const float*)d_in[35];
    const float* fn_ln_b  = (const float*)d_in[36];

    char* ws = (char*)d_ws;
    size_t off = 0;
    auto alloc = [&](size_t bytes) -> char* {
        char* p = ws + off;
        off = (off + bytes + 255) & ~(size_t)255;
        return p;
    };

    // bf16 weight copies (transposed: [N,K], k-contiguous)
    __bf16* wb_f1w1 = (__bf16*)alloc((size_t)D_MODEL * D_FF * 2);
    __bf16* wb_f1w2 = (__bf16*)alloc((size_t)D_FF * D_MODEL * 2);
    __bf16* wb_wq   = (__bf16*)alloc((size_t)D_MODEL * D_MODEL * 2);
    __bf16* wb_wk   = (__bf16*)alloc((size_t)D_MODEL * D_MODEL * 2);
    __bf16* wb_wv   = (__bf16*)alloc((size_t)D_MODEL * D_MODEL * 2);
    __bf16* wb_wo   = (__bf16*)alloc((size_t)D_MODEL * D_MODEL * 2);
    __bf16* wb_pw1  = (__bf16*)alloc((size_t)D_MODEL * 2 * D_MODEL * 2);
    __bf16* wb_pw2  = (__bf16*)alloc((size_t)D_MODEL * D_MODEL * 2);
    __bf16* wb_f2w1 = (__bf16*)alloc((size_t)D_MODEL * D_FF * 2);
    __bf16* wb_f2w2 = (__bf16*)alloc((size_t)D_FF * D_MODEL * 2);
    // activations
    __bf16* lnb   = (__bf16*)alloc((size_t)NTOK * D_MODEL * 2);
    __bf16* hmid  = (__bf16*)alloc((size_t)NTOK * D_FF * 2);
    __bf16* qbuf  = (__bf16*)alloc((size_t)NTOK * D_MODEL * 2);
    __bf16* kbuf  = (__bf16*)alloc((size_t)NTOK * D_MODEL * 2);
    __bf16* vbuf  = (__bf16*)alloc((size_t)NTOK * D_MODEL * 2);
    __bf16* vt    = (__bf16*)alloc((size_t)BATCH * N_HEADS * DKH * SEQ * 2);
    __bf16* ctxb  = (__bf16*)alloc((size_t)NTOK * D_MODEL * 2);
    __bf16* cmb   = (__bf16*)alloc((size_t)NTOK * D_MODEL * 2);
    float* xa     = (float*)alloc((size_t)NTOK * D_MODEL * 4);
    float* xb     = (float*)alloc((size_t)NTOK * D_MODEL * 4);
    float* pw1o   = (float*)alloc((size_t)NTOK * 2 * D_MODEL * 4);
    float* gluo   = (float*)alloc((size_t)NTOK * D_MODEL * 4);

    auto cvt = [&](const float* s, __bf16* d, int K, int N) {
        f2bf_t_k<<<(K * N + 255) / 256, 256, 0, stream>>>(s, d, K, N);
    };
    cvt(f1_w1, wb_f1w1, D_MODEL, D_FF);
    cvt(f1_w2, wb_f1w2, D_FF, D_MODEL);
    cvt(wq, wb_wq, D_MODEL, D_MODEL);
    cvt(wk, wb_wk, D_MODEL, D_MODEL);
    cvt(wv, wb_wv, D_MODEL, D_MODEL);
    cvt(wo, wb_wo, D_MODEL, D_MODEL);
    cvt(pw1_w, wb_pw1, D_MODEL, 2 * D_MODEL);
    cvt(pw2_w, wb_pw2, D_MODEL, D_MODEL);
    cvt(f2_w1, wb_f2w1, D_MODEL, D_FF);
    cvt(f2_w2, wb_f2w2, D_FF, D_MODEL);

    const dim3 blk(256);
    const int lnGrid = NTOK / 8;
    const int ewGrid = (NTOK * D_MODEL) / 256;

    // ---- FFN1: x2 = x + 0.5 * (silu(LN(x)W1+b1) W2 + b2) -> xa
    layernorm_k<0><<<lnGrid, blk, 0, stream>>>(x, f1_ln_g, f1_ln_b, lnb, nullptr);
    gemm_bf16<1><<<dim3(D_FF / 128, NTOK / 128), blk, 0, stream>>>(
        lnb, wb_f1w1, f1_b1, nullptr, nullptr, hmid, NTOK, D_FF, D_MODEL);
    gemm_bf16<2><<<dim3(D_MODEL / 128, NTOK / 128), blk, 0, stream>>>(
        hmid, wb_f1w2, f1_b2, x, xa, nullptr, NTOK, D_MODEL, D_FF);

    // ---- MHSA: x3 = x2 + (attn(LN(x2)) Wo + bo) -> xb
    layernorm_k<0><<<lnGrid, blk, 0, stream>>>(xa, mh_ln_g, mh_ln_b, lnb, nullptr);
    gemm_bf16<3><<<dim3(D_MODEL / 128, NTOK / 128), blk, 0, stream>>>(
        lnb, wb_wq, bq, nullptr, nullptr, qbuf, NTOK, D_MODEL, D_MODEL);
    gemm_bf16<3><<<dim3(D_MODEL / 128, NTOK / 128), blk, 0, stream>>>(
        lnb, wb_wk, bk, nullptr, nullptr, kbuf, NTOK, D_MODEL, D_MODEL);
    gemm_bf16<3><<<dim3(D_MODEL / 128, NTOK / 128), blk, 0, stream>>>(
        lnb, wb_wv, bv, nullptr, nullptr, vbuf, NTOK, D_MODEL, D_MODEL);
    transpose_v_k<<<ewGrid, blk, 0, stream>>>(vbuf, vt);
    attention_k<<<dim3(SEQ / 16, N_HEADS, BATCH), 32, 0, stream>>>(qbuf, kbuf, vt, ctxb);
    gemm_bf16<4><<<dim3(D_MODEL / 128, NTOK / 128), blk, 0, stream>>>(
        ctxb, wb_wo, bo, xa, xb, nullptr, NTOK, D_MODEL, D_MODEL);

    // ---- Conv module: x4 = x3 + pw2(silu(BN(dwconv(GLU(LN(x3)pw1+b))))) -> xa
    layernorm_k<0><<<lnGrid, blk, 0, stream>>>(xb, cv_ln_g, cv_ln_b, lnb, nullptr);
    gemm_bf16<0><<<dim3((2 * D_MODEL) / 128, NTOK / 128), blk, 0, stream>>>(
        lnb, wb_pw1, pw1_b, nullptr, pw1o, nullptr, NTOK, 2 * D_MODEL, D_MODEL);
    glu_k<<<ewGrid, blk, 0, stream>>>(pw1o, gluo);
    dwconv_k<<<ewGrid, blk, 0, stream>>>(gluo, dw_w, dw_b, bn_g, bn_b, bn_mean, bn_var, cmb);
    gemm_bf16<4><<<dim3(D_MODEL / 128, NTOK / 128), blk, 0, stream>>>(
        cmb, wb_pw2, pw2_b, xb, xa, nullptr, NTOK, D_MODEL, D_MODEL);

    // ---- FFN2: x5 = x4 + 0.5 * (silu(LN(x4)W1+b1) W2 + b2) -> xb
    layernorm_k<0><<<lnGrid, blk, 0, stream>>>(xa, f2_ln_g, f2_ln_b, lnb, nullptr);
    gemm_bf16<1><<<dim3(D_FF / 128, NTOK / 128), blk, 0, stream>>>(
        lnb, wb_f2w1, f2_b1, nullptr, nullptr, hmid, NTOK, D_FF, D_MODEL);
    gemm_bf16<2><<<dim3(D_MODEL / 128, NTOK / 128), blk, 0, stream>>>(
        hmid, wb_f2w2, f2_b2, xa, xb, nullptr, NTOK, D_MODEL, D_FF);

    // ---- Final LN -> d_out (f32)
    layernorm_k<1><<<lnGrid, blk, 0, stream>>>(xb, fn_ln_g, fn_ln_b, nullptr, (float*)d_out);
}
